// GRUmodel_71244917506071
// MI455X (gfx1250) — compile-verified, matrix operand order checked by reading, not run on hardware
//
#include <hip/hip_runtime.h>
#include <stdint.h>

// ---------------- problem constants ----------------
#define B_   128
#define T_   256
#define IN_  256
#define H_   512
#define G_   1536            // 3*H
#define MT_  ((B_*T_)/16)    // 2048 m-tiles (m = t*128 + b)
#define GT_  (G_/16)         // 96 gate tiles
#define LDSW (H_ + 8)        // padded LDS row (1040B stride -> conflict-free b128)

typedef __attribute__((ext_vector_type(16))) __bf16 v16bf;
typedef __attribute__((ext_vector_type(8)))  float  v8f;

union Frag16 { v16bf v; uint4 u[2]; };
union FragCf { v8f v; float4 q[2]; };

__device__ __forceinline__ unsigned short f2bf_bits(float f) {
  unsigned int u = __float_as_uint(f);
  u += 0x7fffu + ((u >> 16) & 1u);          // round-to-nearest-even
  return (unsigned short)(u >> 16);
}

// Row-major 16x32 bf16 fragment load (A matrices).
// ISA 16-bit layout: lane L<16 holds row L, K {k0..k0+7, k0+16..k0+23};
// lane L+16 holds row L, K {k0+8..k0+15, k0+24..k0+31}.
__device__ __forceinline__ Frag16 load_frag(const unsigned short* __restrict__ base,
                                            int row0, int k0, int ldk, int lane) {
  Frag16 f;
  const unsigned short* p =
      base + (size_t)(row0 + (lane & 15)) * ldk + k0 + ((lane >> 4) << 3);
  f.u[0] = *(const uint4*)p;
  f.u[1] = *(const uint4*)(p + 16);
  return f;
}

// Packed-fragment load: weights pre-tiled as Wpk[gt][kk][lane][16] bf16 so a
// wave reads one contiguous 1KB burst per fragment.
__device__ __forceinline__ Frag16 load_frag_pk(const unsigned short* __restrict__ base,
                                               int gt, int kk, int K32, int lane) {
  const unsigned short* p = base + (((size_t)gt * K32 + kk) * 32 + lane) * 16;
  Frag16 f;
  f.u[0] = *(const uint4*)p;
  f.u[1] = *(const uint4*)(p + 8);
  return f;
}

// ---------------- conversion / packing kernels ----------------
__global__ void k_pack_w(const float* __restrict__ W,
                         unsigned short* __restrict__ out,
                         int K, int total) {   // total = (Grows/16)*(K/32)*32
  int id = blockIdx.x * blockDim.x + threadIdx.x;
  if (id >= total) return;
  const int K32  = K >> 5;
  const int lane = id & 31;
  const int kk   = (id >> 5) % K32;
  const int gt   = id / (32 * K32);
  const int g     = gt * 16 + (lane & 15);
  const int kbase = kk * 32 + ((lane >> 4) << 3);
  const float* src = W + (size_t)g * K + kbase;
  unsigned short* dst = out + (size_t)id * 16;
#pragma unroll
  for (int c = 0; c < 8; ++c) {
    dst[c]     = f2bf_bits(src[c]);        // K {kbase..kbase+7}
    dst[8 + c] = f2bf_bits(src[16 + c]);   // K {kbase+16..kbase+23}
  }
}

// x: (B,T,IN) f32  ->  xb: (T,B,IN) bf16   (m = t*B + b row-major)
__global__ void k_cvt_x(const float* __restrict__ x, unsigned short* __restrict__ xb) {
  int idx = blockIdx.x * blockDim.x + threadIdx.x;   // over B_*T_*IN_
  int i  = idx & (IN_ - 1);
  int mb = idx >> 8;          // /IN_
  int b  = mb & (B_ - 1);
  int t  = mb >> 7;           // /B_
  xb[idx] = f2bf_bits(x[((size_t)b * T_ + t) * IN_ + i]);
}

// ---------------- xg = X * W^T + b_ih  (tiled-fragment output) --------------
// Double-buffered K-loop: next iteration's fragments are fetched into
// distinct registers before the current WMMAs issue.
__global__ __launch_bounds__(128) void k_gemm_xg(
    const unsigned short* __restrict__ X,
    const unsigned short* __restrict__ Wpk,
    const float* __restrict__ bias,
    float* __restrict__ xg, int K) {
  const int lane  = threadIdx.x & 31;
  const int wv    = threadIdx.x >> 5;           // 0..3
  const int mtile = blockIdx.x;                 // 0..2047
  const int gt0   = (blockIdx.y * 4 + wv) * 4;  // 4 gate tiles per wave
  const int K32   = K >> 5;
  v8f z = {0.f,0.f,0.f,0.f,0.f,0.f,0.f,0.f};
  v8f acc[4] = {z, z, z, z};

  Frag16 a = load_frag(X, mtile * 16, 0, K, lane);
  Frag16 b[4];
#pragma unroll
  for (int jj = 0; jj < 4; ++jj) b[jj] = load_frag_pk(Wpk, gt0 + jj, 0, K32, lane);

#pragma unroll 2
  for (int kk = 0; kk < K32; ++kk) {
    const int kn = (kk + 1 < K32) ? kk + 1 : kk;
    Frag16 a2 = load_frag(X, mtile * 16, kn * 32, K, lane);
    Frag16 b2[4];
#pragma unroll
    for (int jj = 0; jj < 4; ++jj) b2[jj] = load_frag_pk(Wpk, gt0 + jj, kn, K32, lane);
#pragma unroll
    for (int jj = 0; jj < 4; ++jj)
      acc[jj] = __builtin_amdgcn_wmma_f32_16x16x32_bf16(
          false, a.v, false, b[jj].v, (short)0, acc[jj], false, false);
    a = a2;
#pragma unroll
    for (int jj = 0; jj < 4; ++jj) b[jj] = b2[jj];
  }

#pragma unroll
  for (int jj = 0; jj < 4; ++jj) {
    float bi = bias[(gt0 + jj) * 16 + (lane & 15)];
    FragCf r; r.v = acc[jj];
    float4 lo = r.q[0], hi = r.q[1];
    lo.x += bi; lo.y += bi; lo.z += bi; lo.w += bi;
    hi.x += bi; hi.y += bi; hi.z += bi; hi.w += bi;
    float* dst = xg + (((size_t)mtile * GT_ + gt0 + jj) * 32 + lane) * 8;
    *(float4*)dst = lo;
    *(float4*)(dst + 4) = hi;
  }
}

// ---------------- GRU recurrence (persistent over T, no inter-WG sync) -----
// 8 blocks x 1024 threads. Block bg owns batch rows [16*bg,16*bg+16).
// Wave w owns hidden columns [16w,16w+16): gate tiles r=w, z=32+w, n=64+w,
// so gating is wave-local. h kept fp32 in registers (C-fragment layout),
// bf16 copy double-buffered in (bank-skewed) LDS for everyone's A-fragments.
__global__ __launch_bounds__(1024) void k_gru_rec(
    const float* __restrict__ xg,             // tiled [mtile][gt][lane][8]
    const unsigned short* __restrict__ Wpk,   // packed (G_, H_) fragments
    const float* __restrict__ bhh,            // (G_,)
    unsigned short* __restrict__ hout,        // store_all: (T*B,H); else (B,H)
    int store_all) {
  __shared__ unsigned short hbuf[2][16][LDSW];   // ~32.5 KB, skewed rows
  const int lane = threadIdx.x & 31;
  const int wv   = threadIdx.x >> 5;    // 0..31
  const int bg   = blockIdx.x;          // 0..7
  const int col  = lane & 15;
  const int j    = wv * 16 + col;       // hidden column owned by this thread
  const int mhi  = (lane >> 4) << 3;    // accumulator row base: 0 or 8
  const int K32  = H_ >> 5;             // 16

  for (int idx = threadIdx.x; idx < 2 * 16 * LDSW; idx += 1024)
    ((unsigned short*)hbuf)[idx] = 0;

  v8f h = {0.f,0.f,0.f,0.f,0.f,0.f,0.f,0.f};
  const float bhr = bhh[j];
  const float bhz = bhh[H_ + j];
  const float bhn = bhh[2 * H_ + j];
  __syncthreads();

  for (int t = 0; t < T_; ++t) {
    const int cur = t & 1, nxt = cur ^ 1;

    // xg fragments for this (t, bg) tile — issue early, consumed after GEMM
    const float* xb = xg + (size_t)(t * 8 + bg) * GT_ * 256;  // 32 lanes * 8 f32
    const float* pr = xb + ((size_t)(     wv) * 32 + lane) * 8;
    const float* pz = xb + ((size_t)(32 + wv) * 32 + lane) * 8;
    const float* pn = xb + ((size_t)(64 + wv) * 32 + lane) * 8;
    FragCf xR, xZ, xN;
    xR.q[0] = *(const float4*)pr; xR.q[1] = *(const float4*)(pr + 4);
    xZ.q[0] = *(const float4*)pz; xZ.q[1] = *(const float4*)(pz + 4);
    xN.q[0] = *(const float4*)pn; xN.q[1] = *(const float4*)(pn + 4);

    v8f z8 = {0.f,0.f,0.f,0.f,0.f,0.f,0.f,0.f};
    v8f aR = z8, aZ = z8, aN = z8;
    const unsigned short* pa =
        &hbuf[cur][0][0] + (size_t)col * LDSW + ((lane >> 4) << 3);

    Frag16 a, br, bz, bn;
    a.u[0] = *(const uint4*)pa;
    a.u[1] = *(const uint4*)(pa + 16);
    br = load_frag_pk(Wpk,      wv, 0, K32, lane);
    bz = load_frag_pk(Wpk, 32 + wv, 0, K32, lane);
    bn = load_frag_pk(Wpk, 64 + wv, 0, K32, lane);

#pragma unroll 2
    for (int kk = 0; kk < K32; ++kk) {
      const int kn = (kk + 1 < K32) ? kk + 1 : kk;
      Frag16 a2, br2, bz2, bn2;
      const unsigned short* p2 = pa + kn * 32;
      a2.u[0] = *(const uint4*)p2;
      a2.u[1] = *(const uint4*)(p2 + 16);
      br2 = load_frag_pk(Wpk,      wv, kn, K32, lane);
      bz2 = load_frag_pk(Wpk, 32 + wv, kn, K32, lane);
      bn2 = load_frag_pk(Wpk, 64 + wv, kn, K32, lane);
      aR = __builtin_amdgcn_wmma_f32_16x16x32_bf16(false, a.v, false, br.v,
                                                   (short)0, aR, false, false);
      aZ = __builtin_amdgcn_wmma_f32_16x16x32_bf16(false, a.v, false, bz.v,
                                                   (short)0, aZ, false, false);
      aN = __builtin_amdgcn_wmma_f32_16x16x32_bf16(false, a.v, false, bn.v,
                                                   (short)0, aN, false, false);
      a = a2; br = br2; bz = bz2; bn = bn2;
    }

    unsigned short hb16[8];
#pragma unroll
    for (int c = 0; c < 8; ++c) {
      float rr = 1.f / (1.f + __expf(-(xR.v[c] + aR[c] + bhr)));
      float zz = 1.f / (1.f + __expf(-(xZ.v[c] + aZ[c] + bhz)));
      float nn = tanhf(xN.v[c] + rr * (aN[c] + bhn));
      h[c] = (1.f - zz) * nn + zz * h[c];
      hb16[c] = f2bf_bits(h[c]);
    }
#pragma unroll
    for (int c = 0; c < 8; ++c)
      hbuf[nxt][mhi + c][j] = hb16[c];

    if (store_all) {
      unsigned short* o = hout + ((size_t)t * B_ + bg * 16 + mhi) * H_ + j;
#pragma unroll
      for (int c = 0; c < 8; ++c) o[(size_t)c * H_] = hb16[c];
    } else if (t == T_ - 1) {
      unsigned short* o = hout + (size_t)(bg * 16 + mhi) * H_ + j;
#pragma unroll
      for (int c = 0; c < 8; ++c) o[(size_t)c * H_] = hb16[c];
    }
    __syncthreads();   // single barrier per step (double-buffered h)
  }
}

// ---------------- output projection: y = h_last * W_out^T + b_out ----------
__global__ __launch_bounds__(128) void k_outproj(
    const unsigned short* __restrict__ hlast,  // (B_, H_) bf16
    const unsigned short* __restrict__ Wpk,    // packed (64, H_) fragments
    const float* __restrict__ bout,
    float* __restrict__ y) {
  const int lane  = threadIdx.x & 31;
  const int wv    = threadIdx.x >> 5;   // ntile 0..3
  const int mtile = blockIdx.x;         // 0..7
  const int K32   = H_ >> 5;
  v8f acc = {0.f,0.f,0.f,0.f,0.f,0.f,0.f,0.f};
  for (int kk = 0; kk < K32; ++kk) {
    Frag16 a = load_frag(hlast, mtile * 16, kk * 32, H_, lane);
    Frag16 b = load_frag_pk(Wpk, wv, kk, K32, lane);
    acc = __builtin_amdgcn_wmma_f32_16x16x32_bf16(false, a.v, false, b.v,
                                                  (short)0, acc, false, false);
  }
  const int n  = wv * 16 + (lane & 15);
  const float bi = bout[n];
  const int mhi = (lane >> 4) << 3;
#pragma unroll
  for (int c = 0; c < 8; ++c)
    y[(size_t)(mtile * 16 + mhi + c) * 64 + n] = acc[c] + bi;
}

// ---------------- host orchestration ----------------
static inline size_t alignup(size_t x) { return (x + 255) & ~(size_t)255; }

extern "C" void kernel_launch(void* const* d_in, const int* in_sizes, int n_in,
                              void* d_out, int out_size, void* d_ws, size_t ws_size,
                              hipStream_t stream) {
  (void)in_sizes; (void)n_in; (void)out_size; (void)ws_size;
  const float* x     = (const float*)d_in[0];
  const float* Wih0  = (const float*)d_in[1];
  const float* Whh0  = (const float*)d_in[2];
  const float* bih0  = (const float*)d_in[3];
  const float* bhh0  = (const float*)d_in[4];
  const float* Wih1  = (const float*)d_in[5];
  const float* Whh1  = (const float*)d_in[6];
  const float* bih1  = (const float*)d_in[7];
  const float* bhh1  = (const float*)d_in[8];
  const float* Wout  = (const float*)d_in[9];
  const float* bout  = (const float*)d_in[10];

  char* ws = (char*)d_ws;
  size_t off = 0;
  unsigned short* XB    = (unsigned short*)(ws + off); off += alignup((size_t)B_*T_*IN_*2);
  unsigned short* WIH0p = (unsigned short*)(ws + off); off += alignup((size_t)G_*IN_*2);
  unsigned short* WHH0p = (unsigned short*)(ws + off); off += alignup((size_t)G_*H_*2);
  unsigned short* WIH1p = (unsigned short*)(ws + off); off += alignup((size_t)G_*H_*2);
  unsigned short* WHH1p = (unsigned short*)(ws + off); off += alignup((size_t)G_*H_*2);
  unsigned short* WOUTp = (unsigned short*)(ws + off); off += alignup((size_t)64*H_*2);
  float*          XG    = (float*)        (ws + off); off += alignup((size_t)MT_*GT_*32*8*4);
  unsigned short* HSEQ0 = (unsigned short*)(ws + off); off += alignup((size_t)B_*T_*H_*2);
  unsigned short* HLAST = (unsigned short*)(ws + off); off += alignup((size_t)B_*H_*2);

  // 1) convert x (transposed to (T,B,IN)) and pack all weights to bf16 tiles
  k_cvt_x<<<(B_*T_*IN_ + 255) / 256, 256, 0, stream>>>(x, XB);
  {
    int tot;
    tot = (G_/16)*(IN_/32)*32;
    k_pack_w<<<(tot + 255) / 256, 256, 0, stream>>>(Wih0, WIH0p, IN_, tot);
    tot = (G_/16)*(H_/32)*32;
    k_pack_w<<<(tot + 255) / 256, 256, 0, stream>>>(Whh0, WHH0p, H_, tot);
    k_pack_w<<<(tot + 255) / 256, 256, 0, stream>>>(Wih1, WIH1p, H_, tot);
    k_pack_w<<<(tot + 255) / 256, 256, 0, stream>>>(Whh1, WHH1p, H_, tot);
    tot = (64/16)*(H_/32)*32;
    k_pack_w<<<(tot + 255) / 256, 256, 0, stream>>>(Wout, WOUTp, H_, tot);
  }

  // 2) layer 0: xg0 = x*Wih0^T + bih0, then recurrence -> HSEQ0
  k_gemm_xg<<<dim3(MT_, 6), 128, 0, stream>>>(XB, WIH0p, bih0, XG, IN_);
  k_gru_rec<<<8, 1024, 0, stream>>>(XG, WHH0p, bhh0, HSEQ0, 1);

  // 3) layer 1: xg1 = h0*Wih1^T + bih1 (reuses XG), recurrence -> HLAST only
  k_gemm_xg<<<dim3(MT_, 6), 128, 0, stream>>>(HSEQ0, WIH1p, bih1, XG, H_);
  k_gru_rec<<<8, 1024, 0, stream>>>(XG, WHH1p, bhh1, HLAST, 0);

  // 4) output projection
  k_outproj<<<8, 128, 0, stream>>>(HLAST, WOUTp, bout, (float*)d_out);
}
